// KPConv_17712445129349
// MI455X (gfx1250) — matrix operand order repaired and checked by reading, not compiled, not run on hardware
//
#include <hip/hip_runtime.h>
#include <hip/hip_bf16.h>
#include <stdint.h>

typedef __attribute__((ext_vector_type(16))) __bf16 v16bf;
typedef __attribute__((ext_vector_type(8)))  __bf16 v8bf;
typedef __attribute__((ext_vector_type(8)))  float  v8f;
typedef __attribute__((ext_vector_type(4)))  float  v4f;

union U16 { v16bf v; v8bf h[2]; };

#define M_PTS   50000
#define N_PTS   50000
#define H_NB    32
#define C_IN    128
#define C_OUT   128
#define K_PTS   15
#define KPAD    16
#define QT      16          // queries per block
#define SIG_INV 0.5f        // 1/SIGMA

// LDS layout (bytes):
//   kp_s  : [0, 192)        48 f32 (15x3 kernel points, padded)
//   qp_s  : [192, 384)      48 f32 (16x3 query points)
//   Ft    : [384, +131072)  16 q * 128 c * 32 h  bf16   (B fragments: h contiguous)
//   Il    : [+0, +16384)    16 q * 16 k * 32 h   bf16   (A fragments: h contiguous)
//   WF    : [+0, +65536)    16 q * 2048 (c*16+k) bf16   (stage-2 A, j contiguous)
#define OFF_QP  192
#define OFF_FT  384
#define OFF_IL  (OFF_FT + QT*C_IN*H_NB*2)           // 384 + 131072
#define OFF_WF  (OFF_IL + QT*KPAD*H_NB*2)           // + 16384
#define LDS_BYTES (OFF_WF + QT*C_IN*KPAD*2)         // + 65536 = 213376

// ---------------------------------------------------------------------------
// Prep kernel: Wpack[d][c*16+k] = bf16(weights[k][c][d]),  k==15 -> 0
// Transposed layout (j contiguous per output channel d) so stage-2 B
// fragments are two contiguous 16B loads per lane, straight from L2.
// ---------------------------------------------------------------------------
__global__ void kp_pack_weights(const float* __restrict__ w,
                                __bf16* __restrict__ wp) {
  const int d = blockIdx.x;                       // 0..127
  for (int j = threadIdx.x; j < C_IN * KPAD; j += blockDim.x) {
    const int c = j >> 4, k = j & 15;
    const float v = (k < K_PTS) ? w[(k * C_IN + c) * C_OUT + d] : 0.0f;
    wp[d * (C_IN * KPAD) + j] = (__bf16)v;
  }
}

// ---------------------------------------------------------------------------
// Main fused KPConv kernel: 256 threads (8 wave32), 16 queries per block.
// ---------------------------------------------------------------------------
__global__ void __launch_bounds__(256) kpconv_main(
    const float* __restrict__ q_pts, const float* __restrict__ s_pts,
    const float* __restrict__ s_feats, const int* __restrict__ inds,
    const float* __restrict__ kpts, const __bf16* __restrict__ wp,
    float* __restrict__ out)
{
  extern __shared__ char smem[];
  float*  kp_s = (float*)smem;
  float*  qp_s = (float*)(smem + OFF_QP);
  __bf16* Ft   = (__bf16*)(smem + OFF_FT);
  __bf16* Il   = (__bf16*)(smem + OFF_IL);
  __bf16* WF   = (__bf16*)(smem + OFF_WF);

  const int tid   = threadIdx.x;
  const int qbase = blockIdx.x * QT;

  if (tid < K_PTS * 3) kp_s[tid] = kpts[tid];
  if (tid < QT * 3)    qp_s[tid] = q_pts[qbase * 3 + tid];
  __syncthreads();

  // ---------------- Phase A: gather feats (bf16) + influences ----------------
  // One thread per (query, neighbor-pair): 16 q * 16 pairs = 256 tasks.
  {
    const int q  = tid >> 4;
    const int hp = tid & 15;
    const int i0 = inds[(qbase + q) * H_NB + 2 * hp];
    const int i1 = inds[(qbase + q) * H_NB + 2 * hp + 1];
    const bool v0 = (unsigned)i0 < (unsigned)N_PTS;
    const bool v1 = (unsigned)i1 < (unsigned)N_PTS;
    const float m0 = v0 ? 1.0f : 0.0f;
    const float m1 = v1 ? 1.0f : 0.0f;
    const float* f0 = s_feats + (size_t)(v0 ? i0 : 0) * C_IN;
    const float* f1 = s_feats + (size_t)(v1 ? i1 : 0) * C_IN;

    // Feature gather: Ft[q][c][2hp..2hp+1], packed pair store per channel.
    uint32_t* fdst = (uint32_t*)(Ft + q * (C_IN * H_NB) + 2 * hp);
    #pragma unroll 4
    for (int c = 0; c < C_IN; c += 4) {
      v4f a = *(const v4f*)(f0 + c);
      v4f b = *(const v4f*)(f1 + c);
      #pragma unroll
      for (int u = 0; u < 4; ++u) {
        union { uint32_t u32; __bf16 h2[2]; } p;
        p.h2[0] = (__bf16)(a[u] * m0);
        p.h2[1] = (__bf16)(b[u] * m1);
        fdst[(c + u) * (H_NB / 2)] = p.u32;
      }
    }

    // Influences: Il[q][k][2hp..2hp+1]; k=15 row zeroed (pads A to 16 rows).
    const float qx = qp_s[q * 3 + 0], qy = qp_s[q * 3 + 1], qz = qp_s[q * 3 + 2];
    float p0x = 1.0e6f, p0y = 1.0e6f, p0z = 1.0e6f;
    float p1x = 1.0e6f, p1y = 1.0e6f, p1z = 1.0e6f;
    if (v0) { p0x = s_pts[i0*3] - qx; p0y = s_pts[i0*3+1] - qy; p0z = s_pts[i0*3+2] - qz; }
    if (v1) { p1x = s_pts[i1*3] - qx; p1y = s_pts[i1*3+1] - qy; p1z = s_pts[i1*3+2] - qz; }

    uint32_t* idst = (uint32_t*)(Il + q * (KPAD * H_NB) + 2 * hp);
    #pragma unroll
    for (int k = 0; k < KPAD; ++k) {
      float w0 = 0.0f, w1 = 0.0f;
      if (k < K_PTS) {
        const float kx = kp_s[k*3], ky = kp_s[k*3+1], kz = kp_s[k*3+2];
        float dx = p0x - kx, dy = p0y - ky, dz = p0z - kz;
        w0 = fmaxf(0.0f, 1.0f - SIG_INV * sqrtf(dx*dx + dy*dy + dz*dz));
        dx = p1x - kx; dy = p1y - ky; dz = p1z - kz;
        w1 = fmaxf(0.0f, 1.0f - SIG_INV * sqrtf(dx*dx + dy*dy + dz*dz));
      }
      union { uint32_t u32; __bf16 h2[2]; } p;
      p.h2[0] = (__bf16)w0;
      p.h2[1] = (__bf16)w1;
      idst[k * (H_NB / 2)] = p.u32;
    }
  }
  __syncthreads();

  const int lane = tid & 31;
  const int wv   = tid >> 5;      // wave id 0..7
  const int hi   = lane >> 4;     // lane half
  const int lr   = lane & 15;

  // ---------------- Phase B: WF[q] = I_q^T (16x32) x F_q (32x128) ------------
  // One wave handles 2 queries. All 8 B fragments are preloaded so the DS
  // pipe saturates and the 8 independent WMMAs issue back-to-back; converted
  // results are spilled to LDS only at the end (no store->load wait).
  for (int qq = 0; qq < 2; ++qq) {
    const int q = wv * 2 + qq;
    // A fragment: lane half 0 -> h 0..15, half 1 -> h 16..31 (contiguous)
    const __bf16* arow = Il + q * (KPAD * H_NB) + lr * H_NB;
    U16 ua;
    ua.h[0] = *(const v8bf*)(arow + 16 * hi);
    ua.h[1] = *(const v8bf*)(arow + 16 * hi + 8);

    v16bf Bt[8];
    #pragma unroll
    for (int t = 0; t < 8; ++t) {
      const int c = t * 16 + lr;
      // B fragment: half 0 -> h 0..15, half 1 -> h 16..31 (contiguous)
      Bt[t] = *(const v16bf*)(Ft + q * (C_IN * H_NB) + c * H_NB + 16 * hi);
    }

    v8bf ob[8];
    #pragma unroll
    for (int t = 0; t < 8; ++t) {
      v8f acc = {0.f, 0.f, 0.f, 0.f, 0.f, 0.f, 0.f, 0.f};
      acc = __builtin_amdgcn_wmma_f32_16x16x32_bf16(
                false, ua.v, false, Bt[t], (short)0, acc, false, false);
      #pragma unroll
      for (int r = 0; r < 8; ++r) ob[t][r] = (__bf16)acc[r];
    }

    #pragma unroll
    for (int t = 0; t < 8; ++t) {
      const int c = t * 16 + lr;
      // D row r -> k = r + 8*hi at column c: contiguous in WF[q][c*16+k].
      *(v8bf*)(WF + q * (C_IN * KPAD) + c * KPAD + 8 * hi) = ob[t];
    }
  }
  __syncthreads();

  // ---------------- Phase C: out(16x128) = WF(16x2048) @ Wpack(2048x128) -----
  // Each wave owns one 16-channel output tile; 64 k-steps of 32, split into
  // two independent accumulator chains to break the D->C RAW dependency.
  {
    const int dt = wv;
    v8f acc0 = {0.f, 0.f, 0.f, 0.f, 0.f, 0.f, 0.f, 0.f};
    v8f acc1 = {0.f, 0.f, 0.f, 0.f, 0.f, 0.f, 0.f, 0.f};
    const __bf16* a_base = WF + lr * (C_IN * KPAD);
    const __bf16* b_base = wp + (size_t)(dt * 16 + lr) * (C_IN * KPAD) + 16 * hi;
    #pragma unroll 4
    for (int kk = 0; kk < (C_IN * KPAD) / 32; kk += 2) {
      U16 ua0, ua1;
      ua0.h[0] = *(const v8bf*)(a_base + kk * 32 + 16 * hi);
      ua0.h[1] = *(const v8bf*)(a_base + kk * 32 + 16 * hi + 8);
      ua1.h[0] = *(const v8bf*)(a_base + (kk + 1) * 32 + 16 * hi);
      ua1.h[1] = *(const v8bf*)(a_base + (kk + 1) * 32 + 16 * hi + 8);
      const v16bf B0 = *(const v16bf*)(b_base + kk * 32);
      const v16bf B1 = *(const v16bf*)(b_base + (kk + 1) * 32);
      acc0 = __builtin_amdgcn_wmma_f32_16x16x32_bf16(
                 false, ua0.v, false, B0, (short)0, acc0, false, false);
      acc1 = __builtin_amdgcn_wmma_f32_16x16x32_bf16(
                 false, ua1.v, false, B1, (short)0, acc1, false, false);
    }
    const v8f acc = acc0 + acc1;
    float* obase = out + (size_t)qbase * C_OUT + dt * 16 + lr;
    #pragma unroll
    for (int r = 0; r < 8; ++r)
      obase[(size_t)(r + 8 * hi) * C_OUT] = acc[r];
  }
}

// ---------------------------------------------------------------------------
extern "C" void kernel_launch(void* const* d_in, const int* in_sizes, int n_in,
                              void* d_out, int out_size, void* d_ws, size_t ws_size,
                              hipStream_t stream) {
  const float* q_pts   = (const float*)d_in[0];
  const float* s_pts   = (const float*)d_in[1];
  const float* s_feats = (const float*)d_in[2];
  const int*   inds    = (const int*)d_in[3];
  const float* weights = (const float*)d_in[4];
  const float* kpts    = (const float*)d_in[5];
  __bf16* wp = (__bf16*)d_ws;    // 128 * 2048 * 2B = 512 KB packed weights

  (void)hipFuncSetAttribute((const void*)kpconv_main,
                            hipFuncAttributeMaxDynamicSharedMemorySize,
                            (int)LDS_BYTES);

  kp_pack_weights<<<C_OUT, 256, 0, stream>>>(weights, wp);
  kpconv_main<<<M_PTS / QT, 256, LDS_BYTES, stream>>>(
      q_pts, s_pts, s_feats, inds, kpts, wp, (float*)d_out);
}
